// TransformerBlock_63617055588840
// MI455X (gfx1250) — compile-verified, matrix operand order checked by reading, not run on hardware
//
#include <hip/hip_runtime.h>
#include <hip/hip_bf16.h>
#include <cmath>

// ---------------- constants ----------------
constexpr int BATCH    = 2;
constexpr int SEQ      = 2048;
constexpr int EMBED    = 1024;
constexpr int HEADS    = 16;
constexpr int HEAD_DIM = 64;
constexpr int HIDDEN   = 4096;
constexpr int TOK      = BATCH * SEQ;     // 4096
constexpr int QKV_N    = 3 * EMBED;       // 3072

// ---------------- types ----------------
typedef __attribute__((ext_vector_type(8)))  __bf16 bf16x8;
typedef __attribute__((ext_vector_type(16))) __bf16 bf16x16;
typedef __attribute__((ext_vector_type(8)))  float  floatx8;

// ---------------- helpers ----------------
__device__ __forceinline__ unsigned short f2bf(float f) {
    unsigned int u = __float_as_uint(f);
    u += 0x7FFFu + ((u >> 16) & 1u);  // round-to-nearest-even
    return (unsigned short)(u >> 16);
}

// CDNA5 async copy: global -> LDS, 16B per lane, tracked by ASYNCcnt.
__device__ __forceinline__ void async_load_b128(unsigned lds_off, const void* gptr) {
    asm volatile("global_load_async_to_lds_b128 %0, %1, off"
                 :: "v"(lds_off), "v"(gptr)
                 : "memory");
}
__device__ __forceinline__ void wait_async0() {
    asm volatile("s_wait_asynccnt 0" ::: "memory");
}
__device__ __forceinline__ unsigned lds_addr(const void* p) {
    return (unsigned)(size_t)p;   // LDS offset lives in low 32 bits of the generic pointer
}

// Load one 16x32 bf16 fragment's per-lane data: 8 elems at p, 8 at p+16 elems.
// Matches CDNA5 16-bit A/B fragment layout (lanes 0-15: K0-7,K16-23; 16-31: K8-15,K24-31)
__device__ __forceinline__ bf16x16 frag_ld(const unsigned short* p) {
    bf16x8 a = *reinterpret_cast<const bf16x8*>(p);
    bf16x8 b = *reinterpret_cast<const bf16x8*>(p + 16);
    return __builtin_shufflevector(a, b, 0,1,2,3,4,5,6,7,8,9,10,11,12,13,14,15);
}

__device__ __forceinline__ floatx8 wmma_bf16(bf16x16 a, bf16x16 b, floatx8 c) {
    return __builtin_amdgcn_wmma_f32_16x16x32_bf16(false, a, false, b, (short)0, c, false, false);
}

__device__ __forceinline__ float red_max16(float v) {
    #pragma unroll
    for (int mk = 1; mk < 16; mk <<= 1) v = fmaxf(v, __shfl_xor(v, mk, 32));
    return v;
}
__device__ __forceinline__ float red_sum16(float v) {
    #pragma unroll
    for (int mk = 1; mk < 16; mk <<= 1) v += __shfl_xor(v, mk, 32);
    return v;
}

// ---------------- weight transpose + cast: wt[n*K+k] = bf16(w[k*N+n]) ----------------
__global__ __launch_bounds__(256) void transpose_cast(const float* __restrict__ w,
                                                      unsigned short* __restrict__ wt,
                                                      int K, int N) {
    __shared__ float tile[32][33];
    const int n0 = blockIdx.x * 32, k0 = blockIdx.y * 32;
    const int tx = threadIdx.x & 31, ty = threadIdx.x >> 5;
    #pragma unroll
    for (int i = ty; i < 32; i += 8)
        tile[i][tx] = w[(size_t)(k0 + i) * N + n0 + tx];
    __syncthreads();
    #pragma unroll
    for (int i = ty; i < 32; i += 8)
        wt[(size_t)(n0 + i) * K + k0 + tx] = f2bf(tile[tx][i]);
}

// ---------------- LayerNorm -> bf16, one block (256 thr) per token ----------------
__global__ __launch_bounds__(256) void ln_kernel(const float* __restrict__ x,
                                                 const float* __restrict__ g,
                                                 const float* __restrict__ bta,
                                                 unsigned short* __restrict__ out) {
    const size_t t = blockIdx.x;
    const int tid = threadIdx.x;
    const float4 v = reinterpret_cast<const float4*>(x + t * EMBED)[tid];
    float s  = v.x + v.y + v.z + v.w;
    float s2 = v.x*v.x + v.y*v.y + v.z*v.z + v.w*v.w;
    #pragma unroll
    for (int mk = 1; mk < 32; mk <<= 1) { s += __shfl_xor(s, mk, 32); s2 += __shfl_xor(s2, mk, 32); }
    __shared__ float ss[8], ss2[8];
    if ((tid & 31) == 0) { ss[tid >> 5] = s; ss2[tid >> 5] = s2; }
    __syncthreads();
    float ts = 0.f, ts2 = 0.f;
    #pragma unroll
    for (int i = 0; i < 8; ++i) { ts += ss[i]; ts2 += ss2[i]; }
    const float inv = 1.0f / (float)EMBED;
    const float mu  = ts * inv;
    const float var = ts2 * inv - mu * mu;
    const float rstd = rsqrtf(var + 1e-5f);
    const float4 gv = reinterpret_cast<const float4*>(g)[tid];
    const float4 bv = reinterpret_cast<const float4*>(bta)[tid];
    unsigned long long pk =
        (unsigned long long)f2bf((v.x - mu) * rstd * gv.x + bv.x)
      | ((unsigned long long)f2bf((v.y - mu) * rstd * gv.y + bv.y) << 16)
      | ((unsigned long long)f2bf((v.z - mu) * rstd * gv.z + bv.z) << 32)
      | ((unsigned long long)f2bf((v.w - mu) * rstd * gv.w + bv.w) << 48);
    *reinterpret_cast<unsigned long long*>(&out[t * EMBED + tid * 4]) = pk;
}

// ---------------- WMMA GEMM: out = A(M,K) * Bt(N,K)^T + bias, fused epilogues ----------------
enum { MODE_QKV = 0, MODE_PROJ = 1, MODE_GELU = 2, MODE_RES = 3 };

template <int MODE>
__global__ __launch_bounds__(256) void gemm_kernel(
    const unsigned short* __restrict__ A, const unsigned short* __restrict__ Bt,
    const float* __restrict__ bias, const float* __restrict__ resid,
    float* __restrict__ o32, unsigned short* __restrict__ o16,
    unsigned short* __restrict__ oq, unsigned short* __restrict__ ok,
    unsigned short* __restrict__ ovt, int M, int Nn, int K) {
    constexpr int PAD = 40;                 // 32 + 8 bf16 pad -> conflict-free 16B frag loads
    __shared__ unsigned short sA[2][128 * PAD];
    __shared__ unsigned short sB[2][128 * PAD];

    const int tid  = threadIdx.x;
    const int lane = tid & 31, w = tid >> 5;
    const int wm = w & 3, wn = w >> 2;      // 4x2 wave grid -> wave owns 32x64
    const int m0 = blockIdx.y * 128, n0 = blockIdx.x * 128;
    const int r = lane & 15, hf = lane >> 4;

    floatx8 acc[2][4];
    #pragma unroll
    for (int mi = 0; mi < 2; ++mi)
        #pragma unroll
        for (int ni = 0; ni < 4; ++ni)
            acc[mi][ni] = floatx8{0.f,0.f,0.f,0.f,0.f,0.f,0.f,0.f};

    // async prefetch of one 128x32 A tile + 128x32 B tile (512+512 x16B chunks, 4/thread)
    auto issue_tile = [&](int k0, int buf) {
        #pragma unroll
        for (int i = 0; i < 2; ++i) {
            const int idx = tid + 256 * i;
            const int row = idx >> 2, ch = (idx & 3) * 8;
            async_load_b128(lds_addr(&sA[buf][row * PAD + ch]),
                            &A[(size_t)(m0 + row) * K + k0 + ch]);
            async_load_b128(lds_addr(&sB[buf][row * PAD + ch]),
                            &Bt[(size_t)(n0 + row) * K + k0 + ch]);
        }
    };

    const int nk = K >> 5;
    issue_tile(0, 0);
    for (int kt = 0; kt < nk; ++kt) {
        const int buf = kt & 1;
        wait_async0();          // this wave's chunks for tile kt have landed
        __syncthreads();        // whole tile visible; prev reads of other buffer done
        if (kt + 1 < nk) issue_tile((kt + 1) << 5, buf ^ 1);

        // gather ALL fragments first so LDS latency overlaps with WMMA issue
        bf16x16 af[2], bfv[4];
        #pragma unroll
        for (int mi = 0; mi < 2; ++mi)
            af[mi] = frag_ld(&sA[buf][(wm * 32 + mi * 16 + r) * PAD + hf * 8]);
        #pragma unroll
        for (int ni = 0; ni < 4; ++ni)
            bfv[ni] = frag_ld(&sB[buf][(wn * 64 + ni * 16 + r) * PAD + hf * 8]);
        #pragma unroll
        for (int ni = 0; ni < 4; ++ni)
            #pragma unroll
            for (int mi = 0; mi < 2; ++mi)
                acc[mi][ni] = wmma_bf16(af[mi], bfv[ni], acc[mi][ni]);
    }

    #pragma unroll
    for (int mi = 0; mi < 2; ++mi) {
        #pragma unroll
        for (int ni = 0; ni < 4; ++ni) {
            #pragma unroll
            for (int rr = 0; rr < 8; ++rr) {
                const int row = m0 + wm * 32 + mi * 16 + rr + 8 * hf;
                const int col = n0 + wn * 64 + ni * 16 + r;
                const float v = acc[mi][ni][rr] + bias[col];
                if constexpr (MODE == MODE_QKV) {
                    const int which = col >> 10, hh = (col >> 6) & 15, dd = col & 63;
                    const int bb = row >> 11, nn = row & 2047;
                    const unsigned short bv = f2bf(v);
                    const size_t qi = ((size_t)(bb * HEADS + hh) * SEQ + nn) * HEAD_DIM + dd;
                    if (which == 0)      oq[qi] = bv;
                    else if (which == 1) ok[qi] = bv;
                    else ovt[((size_t)(bb * HEADS + hh) * HEAD_DIM + dd) * SEQ + nn] = bv;
                } else if constexpr (MODE == MODE_PROJ || MODE == MODE_RES) {
                    const size_t oi = (size_t)row * Nn + col;
                    o32[oi] = resid[oi] + v;
                } else { // MODE_GELU (exact)
                    const float gg = 0.5f * v * (1.0f + erff(v * 0.70710678118f));
                    o16[(size_t)row * Nn + col] = f2bf(gg);
                }
            }
        }
    }
}

// ---------------- flash attention: q[B,H,N,D], k[B,H,N,D], vt[B,H,D,N] -> out[B,N,C] bf16 ----
__global__ __launch_bounds__(256) void attn_kernel(const unsigned short* __restrict__ q,
                                                   const unsigned short* __restrict__ k,
                                                   const unsigned short* __restrict__ vt,
                                                   unsigned short* __restrict__ out) {
    constexpr int KP = 72;   // 64 + 8 pad
    constexpr int VP = 40;   // 32 + 8 pad
    __shared__ unsigned short sK[2][32 * KP];     // 32 keys x 64 d
    __shared__ unsigned short sV[2][64 * VP];     // 64 d x 32 keys (from vt)
    __shared__ unsigned short sP[8 * 16 * VP];    // per-wave P staging (D-layout -> A-layout)

    const int tid = threadIdx.x, lane = tid & 31, w = tid >> 5;
    const int b = blockIdx.z, h = blockIdx.y, q0 = blockIdx.x * 128;
    const int r = lane & 15, hf = lane >> 4;
    const size_t bh = (size_t)(b * HEADS + h);
    const unsigned short* qbase = q  + (bh * SEQ + q0 + w * 16) * HEAD_DIM;
    const unsigned short* kbase = k  + bh * SEQ * HEAD_DIM;
    const unsigned short* vbase = vt + bh * HEAD_DIM * SEQ;
    unsigned short* pw = &sP[w * 16 * VP];

    bf16x16 aq[2];
    #pragma unroll
    for (int ds = 0; ds < 2; ++ds)
        aq[ds] = frag_ld(&qbase[(size_t)r * HEAD_DIM + ds * 32 + hf * 8]);

    floatx8 accv[4];
    #pragma unroll
    for (int ni = 0; ni < 4; ++ni) accv[ni] = floatx8{0.f,0.f,0.f,0.f,0.f,0.f,0.f,0.f};
    float m[8], l[8];
    #pragma unroll
    for (int rr = 0; rr < 8; ++rr) { m[rr] = -3.0e38f; l[rr] = 0.f; }

    // async prefetch of one K tile (32x64, 256 x16B) + one V tile (64x32, 256 x16B), 1+1/thread
    const int krow = tid >> 3, kch = (tid & 7) * 8;
    const int vrow = tid >> 2, vch = (tid & 3) * 8;
    auto issue_kv = [&](int kt, int buf) {
        async_load_b128(lds_addr(&sK[buf][krow * KP + kch]),
                        &kbase[(size_t)(kt + krow) * HEAD_DIM + kch]);
        async_load_b128(lds_addr(&sV[buf][vrow * VP + vch]),
                        &vbase[(size_t)vrow * SEQ + kt + vch]);
    };

    issue_kv(0, 0);
    for (int kt = 0; kt < SEQ; kt += 32) {
        const int buf = (kt >> 5) & 1;
        wait_async0();
        __syncthreads();
        if (kt + 32 < SEQ) issue_kv(kt + 32, buf ^ 1);

        // gather all K and V fragments up front (V loads overlap S-WMMA + softmax)
        bf16x16 bk[2][2], bv[4];
        #pragma unroll
        for (int nt = 0; nt < 2; ++nt)
            #pragma unroll
            for (int ds = 0; ds < 2; ++ds)
                bk[nt][ds] = frag_ld(&sK[buf][(nt * 16 + r) * KP + ds * 32 + hf * 8]);
        #pragma unroll
        for (int ni = 0; ni < 4; ++ni)
            bv[ni] = frag_ld(&sV[buf][(ni * 16 + r) * VP + hf * 8]);

        // S = q * k^T  (scaled)
        floatx8 S[2];
        #pragma unroll
        for (int nt = 0; nt < 2; ++nt) {
            S[nt] = floatx8{0.f,0.f,0.f,0.f,0.f,0.f,0.f,0.f};
            #pragma unroll
            for (int ds = 0; ds < 2; ++ds)
                S[nt] = wmma_bf16(aq[ds], bk[nt][ds], S[nt]);
            S[nt] = S[nt] * 0.125f;   // 1/sqrt(64)
        }

        // online softmax (rows live in lane-halves, one row per accumulator slot)
        float p0[8], p1[8];
        #pragma unroll
        for (int rr = 0; rr < 8; ++rr) {
            float mx = red_max16(fmaxf(S[0][rr], S[1][rr]));
            const float nm = fmaxf(m[rr], mx);
            const float sc = __expf(m[rr] - nm);
            m[rr] = nm;
            const float e0 = __expf(S[0][rr] - nm);
            const float e1 = __expf(S[1][rr] - nm);
            l[rr] = l[rr] * sc + red_sum16(e0 + e1);
            #pragma unroll
            for (int ni = 0; ni < 4; ++ni) accv[ni][rr] = accv[ni][rr] * sc;
            p0[rr] = e0; p1[rr] = e1;
        }

        // D-layout -> A-layout via per-wave LDS staging
        #pragma unroll
        for (int rr = 0; rr < 8; ++rr) {
            pw[(rr + 8 * hf) * VP + r]      = f2bf(p0[rr]);
            pw[(rr + 8 * hf) * VP + 16 + r] = f2bf(p1[rr]);
        }
        const bf16x16 aP = frag_ld(&pw[r * VP + hf * 8]);
        #pragma unroll
        for (int ni = 0; ni < 4; ++ni)
            accv[ni] = wmma_bf16(aP, bv[ni], accv[ni]);
    }

    #pragma unroll
    for (int rr = 0; rr < 8; ++rr) {
        const float inv = 1.0f / l[rr];
        const int row = q0 + w * 16 + rr + 8 * hf;
        #pragma unroll
        for (int ni = 0; ni < 4; ++ni) {
            const int col = h * HEAD_DIM + ni * 16 + r;
            out[((size_t)(b * SEQ) + row) * EMBED + col] = f2bf(accv[ni][rr] * inv);
        }
    }
}

// ---------------- host launcher ----------------
extern "C" void kernel_launch(void* const* d_in, const int* in_sizes, int n_in,
                              void* d_out, int out_size, void* d_ws, size_t ws_size,
                              hipStream_t stream) {
    const float* x     = (const float*)d_in[0];
    const float* ln1w  = (const float*)d_in[1];
    const float* ln1b  = (const float*)d_in[2];
    const float* ln2w  = (const float*)d_in[3];
    const float* ln2b  = (const float*)d_in[4];
    const float* qkvw  = (const float*)d_in[5];
    const float* qkvb  = (const float*)d_in[6];
    const float* projw = (const float*)d_in[7];
    const float* projb = (const float*)d_in[8];
    const float* fc1w  = (const float*)d_in[9];
    const float* fc1b  = (const float*)d_in[10];
    const float* fc2w  = (const float*)d_in[11];
    const float* fc2b  = (const float*)d_in[12];
    float* out = (float*)d_out;

    char* ws = (char*)d_ws;
    size_t off = 0;
    auto alloc = [&](size_t bytes) -> char* {
        char* p = ws + off;
        off += (bytes + 255) & ~(size_t)255;
        return p;
    };
    unsigned short* wqkvT = (unsigned short*)alloc((size_t)QKV_N  * EMBED  * 2);
    unsigned short* wprjT = (unsigned short*)alloc((size_t)EMBED  * EMBED  * 2);
    unsigned short* wfc1T = (unsigned short*)alloc((size_t)HIDDEN * EMBED  * 2);
    unsigned short* wfc2T = (unsigned short*)alloc((size_t)EMBED  * HIDDEN * 2);
    unsigned short* xln   = (unsigned short*)alloc((size_t)TOK * EMBED * 2);
    unsigned short* qb    = (unsigned short*)alloc((size_t)TOK * EMBED * 2);
    unsigned short* kb    = (unsigned short*)alloc((size_t)TOK * EMBED * 2);
    unsigned short* vtb   = (unsigned short*)alloc((size_t)TOK * EMBED * 2);
    unsigned short* attn  = (unsigned short*)alloc((size_t)TOK * EMBED * 2);
    float*          x1    = (float*)         alloc((size_t)TOK * EMBED * 4);
    unsigned short* hbuf  = (unsigned short*)alloc((size_t)TOK * HIDDEN * 2);

    const dim3 blk(256);

    // bf16-transpose the four weight matrices
    transpose_cast<<<dim3(QKV_N / 32,  EMBED / 32),  blk, 0, stream>>>(qkvw,  wqkvT, EMBED,  QKV_N);
    transpose_cast<<<dim3(EMBED / 32,  EMBED / 32),  blk, 0, stream>>>(projw, wprjT, EMBED,  EMBED);
    transpose_cast<<<dim3(HIDDEN / 32, EMBED / 32),  blk, 0, stream>>>(fc1w,  wfc1T, EMBED,  HIDDEN);
    transpose_cast<<<dim3(EMBED / 32,  HIDDEN / 32), blk, 0, stream>>>(fc2w,  wfc2T, HIDDEN, EMBED);

    // x -> ln1 -> qkv (scattered to q/k/vt)
    ln_kernel<<<dim3(TOK), blk, 0, stream>>>(x, ln1w, ln1b, xln);
    gemm_kernel<MODE_QKV><<<dim3(QKV_N / 128, TOK / 128), blk, 0, stream>>>(
        xln, wqkvT, qkvb, nullptr, nullptr, nullptr, qb, kb, vtb, TOK, QKV_N, EMBED);

    // attention
    attn_kernel<<<dim3(SEQ / 128, HEADS, BATCH), blk, 0, stream>>>(qb, kb, vtb, attn);

    // proj + residual -> x1
    gemm_kernel<MODE_PROJ><<<dim3(EMBED / 128, TOK / 128), blk, 0, stream>>>(
        attn, wprjT, projb, x, x1, nullptr, nullptr, nullptr, nullptr, TOK, EMBED, EMBED);

    // ln2 -> fc1 + gelu -> fc2 + residual -> out
    ln_kernel<<<dim3(TOK), blk, 0, stream>>>(x1, ln2w, ln2b, xln);
    gemm_kernel<MODE_GELU><<<dim3(HIDDEN / 128, TOK / 128), blk, 0, stream>>>(
        xln, wfc1T, fc1b, nullptr, nullptr, hbuf, nullptr, nullptr, nullptr, TOK, HIDDEN, EMBED);
    gemm_kernel<MODE_RES><<<dim3(EMBED / 128, TOK / 128), blk, 0, stream>>>(
        hbuf, wfc2T, fc2b, x1, out, nullptr, nullptr, nullptr, nullptr, TOK, EMBED, HIDDEN);
}
